// SVDMModel_14568529068444
// MI455X (gfx1250) — matrix operand order repaired and checked by reading, not compiled, not run on hardware
//
#include <hip/hip_runtime.h>
#include <stdint.h>

typedef __attribute__((ext_vector_type(16))) _Float16 v16h;
typedef __attribute__((ext_vector_type(8)))  float    v8f;

#define LSZ   64
#define HRSZ  256
#define NPTS  (8 * HRSZ * HRSZ)          // 524288 points
#define MSI_OUT (8 * 4 * LSZ * LSZ)      // 131072 floats (pre_msi)

// ---------------- LDS layout (bytes) ----------------
// [0, 28672)         packed W2 B-fragments : 28 frags x 32 lanes x 8 u32
// [28672, 30720)     packed head B-frags   : 2 frags x 32 lanes x 8 u32
// [30720 + w*9216)   per-wave region:
//    +0     : f-stage  16 x 224 f16 (7168 B)   (also reused as d-stage 16x16 f32)
//    +7168  : f2-stage 16 x 64  f16 (2048 B)
#define LDS_W2   0
#define LDS_WH   28672
#define LDS_WAVE 30720
#define PW       9216
#define LDS_TOTAL (LDS_WAVE + 8 * PW)    // 104448 B per block

union Frag16 { v16h v; float4 f4[2]; _Float16 h[16]; uint32_t u[8]; };
union Frag32 { v8f v; float f[8]; };

__device__ __forceinline__ uint32_t pack2f16(float a, float b) {
  union { _Float16 h[2]; uint32_t u; } x;
  x.h[0] = (_Float16)a; x.h[1] = (_Float16)b;
  return x.u;
}

__global__ void __launch_bounds__(256) mlp_wmma_kernel(
    const float* __restrict__ coords, const float* __restrict__ hrhs,
    const float* __restrict__ w1, const float* __restrict__ b1,
    const float* __restrict__ w2, const float* __restrict__ b2,
    const float* __restrict__ w_spek, const float* __restrict__ b_spek,
    const float* __restrict__ w_spak, const float* __restrict__ b_spak,
    float* __restrict__ out, float* __restrict__ inv_ws)
{
  extern __shared__ char smem[];
  uint32_t* sW2 = (uint32_t*)(smem + LDS_W2);
  uint32_t* sWH = (uint32_t*)(smem + LDS_WH);

  const int tid = threadIdx.x;

  // ---- cooperative weight packing into exact B-fragment layout ----
  // B 32x16 f16 layout: VGPR n, lane L: h[2n]=elem(K=kbase+(n/4)*16+(L/16)*8+2*(n%4), N=L%16)
  for (int u = tid; u < 30 * 256; u += 256) {
    int frag  = u >> 8;       // 0..29
    int r     = u & 255;      // lane*8 + n
    int ln    = r >> 3;
    int n     = r & 7;
    int lhalf = ln >> 4;
    int lN    = ln & 15;
    float h0 = 0.f, h1 = 0.f;
    if (frag < 28) {          // W2: kt = frag/4, nt = frag%4
      int kt = frag >> 2, nt = frag & 3;
      int K0  = kt * 32 + (n >> 2) * 16 + lhalf * 8 + 2 * (n & 3);
      int col = nt * 16 + lN;
      if (col < 50) {
        if (K0     < 200) h0 = w2[K0 * 50 + col];
        if (K0 + 1 < 200) h1 = w2[(K0 + 1) * 50 + col];
      }
      sW2[u] = pack2f16(h0, h1);
    } else {                  // fused head: cols 0..4 = w_spek, 5..8 = w_spak
      int kt = frag - 28;
      int K0 = kt * 32 + (n >> 2) * 16 + lhalf * 8 + 2 * (n & 3);
      int col = lN;
      if (col < 5) {
        if (K0     < 50) h0 = w_spek[K0 * 5 + col];
        if (K0 + 1 < 50) h1 = w_spek[(K0 + 1) * 5 + col];
      } else if (col < 9) {
        if (K0     < 50) h0 = w_spak[K0 * 4 + (col - 5)];
        if (K0 + 1 < 50) h1 = w_spak[(K0 + 1) * 4 + (col - 5)];
      }
      sWH[u - 28 * 256] = pack2f16(h0, h1);
    }
  }
  __syncthreads();

  const int wave  = tid >> 5;
  const int lane  = tid & 31;
  const int lN    = lane & 15;       // column / row-in-tile index
  const int lhalf = lane >> 4;
  const int tileBase = blockIdx.x * 128 + wave * 16;   // 16 points per wave

  _Float16* fst  = (_Float16*)(smem + LDS_WAVE + wave * PW);
  _Float16* f2st = (_Float16*)(smem + LDS_WAVE + wave * PW + 7168);
  float*    dstg = (float*)(smem + LDS_WAVE + wave * PW);  // reuses f-stage

  // ================= GEMM1: [16x2] x [2x200] + b1, relu =================
  Frag16 a1;
  #pragma unroll
  for (int i = 0; i < 8; ++i) a1.u[i] = 0;
  {
    int p = tileBase + lN;
    if (lhalf == 0)  // lanes 0-15 hold K=0,1 in VGPR0; all other K are zero-pad
      a1.u[0] = pack2f16(coords[p * 2], coords[p * 2 + 1]);
  }
  #pragma unroll
  for (int t = 0; t < 13; ++t) {
    Frag16 bfr;
    #pragma unroll
    for (int i = 0; i < 8; ++i) bfr.u[i] = 0;
    int col = t * 16 + lN;
    if (lhalf == 0 && col < 200)
      bfr.u[0] = pack2f16(w1[col], w1[200 + col]);   // rows K=0,1 of W1
    Frag32 c;
    float bias = (col < 200) ? b1[col] : 0.f;
    #pragma unroll
    for (int i = 0; i < 8; ++i) c.f[i] = bias;
    c.v = __builtin_amdgcn_wmma_f32_16x16x32_f16(false, a1.v, false, bfr.v,
                                                 (short)0, c.v, false, false);
    #pragma unroll
    for (int r = 0; r < 8; ++r) {      // D: VGPR r, lane L -> (M=r+8*half, N=L%16)
      int point = r + 8 * lhalf;
      float vv = c.f[r];
      vv = vv > 0.f ? vv : 0.f;        // relu
      fst[point * 224 + col] = (_Float16)vv;
    }
  }
  // zero K-padding cols 208..223 of f-stage
  {
    float4 z = make_float4(0.f, 0.f, 0.f, 0.f);
    *(float4*)((char*)fst + lN * 448 + 416 + lhalf * 16) = z;
  }

  // ===== load A fragments for GEMM2 (A 16x32 f16 layout, 7 K-steps) =====
  Frag16 a2[7];
  #pragma unroll
  for (int kt = 0; kt < 7; ++kt) {
    char* base = (char*)fst + lN * 448 + kt * 64 + lhalf * 16;
    a2[kt].f4[0] = *(float4*)(base);        // K = kt*32 + half*8 + [0..7]
    a2[kt].f4[1] = *(float4*)(base + 32);   // K = kt*32 + 16 + half*8 + [0..7]
  }

  // ============ GEMM2: [16x200] x [200x50] + b2, relu ============
  #pragma unroll
  for (int nt = 0; nt < 4; ++nt) {
    Frag32 c;
    int col = nt * 16 + lN;
    float bias = (col < 50) ? b2[col] : 0.f;
    #pragma unroll
    for (int i = 0; i < 8; ++i) c.f[i] = bias;
    #pragma unroll
    for (int kt = 0; kt < 7; ++kt) {
      Frag16 bfr;
      uint32_t* wsrc = sW2 + (((kt * 4 + nt) * 32 + lane) * 8);
      bfr.f4[0] = *(float4*)(wsrc);
      bfr.f4[1] = *(float4*)(wsrc + 4);
      c.v = __builtin_amdgcn_wmma_f32_16x16x32_f16(false, a2[kt].v, false, bfr.v,
                                                   (short)0, c.v, false, false);
    }
    #pragma unroll
    for (int r = 0; r < 8; ++r) {
      int point = r + 8 * lhalf;
      float vv = c.f[r];
      vv = vv > 0.f ? vv : 0.f;
      f2st[point * 64 + col] = (_Float16)vv;   // cols 50..63 are relu(0)=0
    }
  }

  // ========= fused heads: [16x50(pad64)] x [64x16] (cols 0-4 spek, 5-8 spak) =========
  Frag32 hd;
  {
    float bias = 0.f;
    if (lN < 5)      bias = b_spek[lN];
    else if (lN < 9) bias = b_spak[lN - 5];
    #pragma unroll
    for (int i = 0; i < 8; ++i) hd.f[i] = bias;
    #pragma unroll
    for (int kt = 0; kt < 2; ++kt) {
      Frag16 ah, bh;
      char* base = (char*)f2st + lN * 128 + kt * 64 + lhalf * 16;
      ah.f4[0] = *(float4*)(base);
      ah.f4[1] = *(float4*)(base + 32);
      uint32_t* wsrc = sWH + ((kt * 32 + lane) * 8);
      bh.f4[0] = *(float4*)(wsrc);
      bh.f4[1] = *(float4*)(wsrc + 4);
      hd.v = __builtin_amdgcn_wmma_f32_16x16x32_f16(false, ah.v, false, bh.v,
                                                    (short)0, hd.v, false, false);
    }
  }
  // stage head outputs (16 points x 16 cols f32) for cross-lane epilogue
  #pragma unroll
  for (int r = 0; r < 8; ++r) {
    int point = r + 8 * lhalf;
    dstg[point * 16 + lN] = hd.f[r];
  }

  // ========================= epilogue =========================
  if (lhalf == 0) {
    // lanes 0-15: pre_pan for point lN (coalesced hrhs loads across x)
    int p = tileBase + lN;
    int b = p >> 16;
    int y = (p >> 8) & 255;
    int x = p & 255;
    float pan = dstg[lN * 16 + 4];              // spek[4] * ones-channel
    #pragma unroll
    for (int cch = 0; cch < 4; ++cch)
      pan += hrhs[((b * 4 + cch) << 16) + (y << 8) + x] * dstg[lN * 16 + cch];
    out[MSI_OUT + p] = pan;
  } else {
    // lanes 16-31: INV = M^T M for spak points (y%4==1 && x%4==1)
    int l = lN;
    int p = tileBase + l;
    int y = (p >> 8) & 255;
    int x = p & 255;
    if ((y & 3) == 1 && (x & 3) == 1) {
      int b = p >> 16;
      float v0 = dstg[l * 16 + 5];   // M = [[v0, 0], [v2, v3]] after mask
      float v2 = dstg[l * 16 + 7];
      float v3 = dstg[l * 16 + 8];
      int sidx = (b * 64 + (y >> 2)) * 64 + (x >> 2);
      inv_ws[sidx * 3 + 0] = v0 * v0 + v2 * v2;  // INV00
      inv_ws[sidx * 3 + 1] = v2 * v3;            // INV01
      inv_ws[sidx * 3 + 2] = v3 * v3;            // INV11
    }
  }
}

// ===== kernel 2: normalized Gaussian 15x15 gather -> pre_msi =====
__global__ void __launch_bounds__(256) msi_kernel(
    const float* __restrict__ hrhs, const float* __restrict__ inv_ws,
    float* __restrict__ out)
{
  int id = blockIdx.x * blockDim.x + threadIdx.x;   // one per (b,c,h,w)
  int w  = id & 63;
  int h  = (id >> 6) & 63;
  int cc = (id >> 12) & 3;
  int b  = id >> 14;
  int sidx = (b * 64 + h) * 64 + w;
  float i00 = inv_ws[sidx * 3 + 0];
  float i01 = inv_ws[sidx * 3 + 1];
  float i11 = inv_ws[sidx * 3 + 2];
  const float* src = hrhs + ((b * 4 + cc) << 16);
  int y0 = 4 * h - 6;
  int x0 = 4 * w - 6;
  float sum_e = 0.f, sum_pe = 0.f;
  for (int p = 0; p < 15; ++p) {
    float gp = (float)(p - 7);
    int yy = y0 + p;
    bool yok = (yy >= 0) && (yy < 256);
    #pragma unroll
    for (int q = 0; q < 15; ++q) {
      float gq = (float)(q - 7);
      float t = i00 * gp * gp + 2.f * i01 * gp * gq + i11 * gq * gq;
      float e = __expf(-0.5f * t);
      sum_e += e;                     // normalize over ALL taps (matches ref)
      int xx = x0 + q;
      if (yok && xx >= 0 && xx < 256)
        sum_pe += e * src[(yy << 8) + xx];
    }
  }
  out[id] = sum_pe / sum_e;
}

extern "C" void kernel_launch(void* const* d_in, const int* in_sizes, int n_in,
                              void* d_out, int out_size, void* d_ws, size_t ws_size,
                              hipStream_t stream) {
  const float* coords = (const float*)d_in[0];
  const float* hrhs   = (const float*)d_in[1];
  const float* w1     = (const float*)d_in[2];
  const float* b1     = (const float*)d_in[3];
  const float* w2     = (const float*)d_in[4];
  const float* b2     = (const float*)d_in[5];
  const float* w_spek = (const float*)d_in[6];
  const float* b_spek = (const float*)d_in[7];
  const float* w_spak = (const float*)d_in[8];
  const float* b_spak = (const float*)d_in[9];
  float* out    = (float*)d_out;
  float* inv_ws = (float*)d_ws;   // 32768 * 3 floats = 384 KB

  // kernel 1: 524288 points / (8 waves * 16 pts) = 4096 blocks of 256
  hipLaunchKernelGGL(mlp_wmma_kernel, dim3(4096), dim3(256), LDS_TOTAL, stream,
                     coords, hrhs, w1, b1, w2, b2, w_spek, b_spek, w_spak, b_spak,
                     out, inv_ws);
  // kernel 2: 131072 pre_msi outputs
  hipLaunchKernelGGL(msi_kernel, dim3(512), dim3(256), 0, stream,
                     hrhs, inv_ws, out);
}
